// GATv2WithLogits_9088150798342
// MI455X (gfx1250) — compile-verified, matrix operand order checked by reading, not run on hardware
//
#include <hip/hip_runtime.h>
#include <hip/hip_bf16.h>

typedef __attribute__((ext_vector_type(16))) __bf16 v16bf;
typedef __attribute__((ext_vector_type(8)))  float  v8f;

#define IN_CH   128
#define HEADS   4
#define OUT_CH  16
#define HC      (HEADS * OUT_CH)   // 64
#define KP      (IN_CH / 2)        // 64 packed-bf16 K pairs
#define LDSW_STRIDE (KP + 1)       // pad to break bank-conflict stride
#define NEG_SLOPE 0.2f
#define EPS 1e-16f

// ---- sortable-uint encoding for float atomic max ----
__device__ __forceinline__ unsigned f2s(float f) {
    unsigned u = __float_as_uint(f);
    return (u & 0x80000000u) ? ~u : (u | 0x80000000u);
}
__device__ __forceinline__ float s2f(unsigned s) {
    unsigned u = (s & 0x80000000u) ? (s & 0x7FFFFFFFu) : ~s;
    return __uint_as_float(u);
}
#define SORTABLE_NEG_INF 0x007FFFFFu   // f2s(-inf)

union BfPair { unsigned u; __bf16 h[2]; };
union BfTile { unsigned u[8]; v16bf v; };

// ------------------------------------------------------------------
// Kernel 0: init outputs / segment buffers
// ------------------------------------------------------------------
__global__ __launch_bounds__(256) void gat_init(float* __restrict__ out,
                                                unsigned* __restrict__ seg_max,
                                                float* __restrict__ seg_sum,
                                                int n_nodes) {
    int i = blockIdx.x * blockDim.x + threadIdx.x;
    if (i < n_nodes * OUT_CH) out[i] = 0.0f;
    if (i < n_nodes * HEADS) {
        seg_max[i] = SORTABLE_NEG_INF;
        seg_sum[i] = 0.0f;
    }
}

// ------------------------------------------------------------------
// Kernel 1: x_proj = x @ W via v_wmma_f32_16x16x32_bf16.
// One wave per 16-row tile. W is staged once per block in LDS as
// K-pair-packed bf16 (transposed), shared by all 8 waves.
// A/B VGPR layouts per CDNA5 ISA 7.12.2 (16-bit matrices, wave32).
// ------------------------------------------------------------------
__global__ __launch_bounds__(256) void gat_gemm_wmma(const float* __restrict__ x,
                                                     const float* __restrict__ W,
                                                     float* __restrict__ xp,
                                                     int n_nodes, int n_tiles) {
    // ldsW[n][kk] = packed { bf16 W[2kk][n], bf16 W[2kk+1][n] }
    __shared__ unsigned ldsW[HC * LDSW_STRIDE];
    for (int i = threadIdx.x; i < HC * KP; i += blockDim.x) {
        const int n  = i & (HC - 1);   // fastest index -> coalesced W reads
        const int kk = i >> 6;
        BfPair pk;
        pk.h[0] = (__bf16)W[(size_t)(2 * kk) * HC + n];
        pk.h[1] = (__bf16)W[(size_t)(2 * kk + 1) * HC + n];
        ldsW[n * LDSW_STRIDE + kk] = pk.u;
    }
    __syncthreads();

    const int wave = blockIdx.x * (blockDim.x >> 5) + (threadIdx.x >> 5);
    if (wave >= n_tiles) return;
    const int lane = threadIdx.x & 31;
    const int m16  = lane & 15;
    const int hi   = lane >> 4;

    int row = wave * 16 + m16;
    if (row >= n_nodes) row = n_nodes - 1;   // clamp (stores are guarded)
    const float2* xr = (const float2*)(x + (size_t)row * IN_CH);

    // K-pair offsets of the 8 dwords this lane holds (A and B share pattern):
    // dword v<4: pair = v (+4 if lane>=16); v>=4: pair = 8+(v-4) (+4 if lane>=16)
    int koff2[8];
#pragma unroll
    for (int v = 0; v < 8; ++v)
        koff2[v] = ((v < 4) ? v : (8 + (v - 4))) + hi * 4;

    v8f acc[4] = {v8f{}, v8f{}, v8f{}, v8f{}};

#pragma unroll
    for (int kt2 = 0; kt2 < KP; kt2 += 16) {    // 16 K-pairs = K block of 32
        BfTile A;
#pragma unroll
        for (int v = 0; v < 8; ++v) {
            const float2 p = xr[kt2 + koff2[v]];
            BfPair pk;
            pk.h[0] = (__bf16)p.x;
            pk.h[1] = (__bf16)p.y;
            A.u[v] = pk.u;
        }
#pragma unroll
        for (int nt = 0; nt < 4; ++nt) {
            const int n = nt * 16 + m16;
            BfTile B;
#pragma unroll
            for (int v = 0; v < 8; ++v)
                B.u[v] = ldsW[n * LDSW_STRIDE + kt2 + koff2[v]];
            acc[nt] = __builtin_amdgcn_wmma_f32_16x16x32_bf16(
                false, A.v, false, B.v, (short)0, acc[nt], false, false);
        }
    }

    // D layout: VGPR r -> M = r + 8*hi, N = nt*16 + m16
#pragma unroll
    for (int nt = 0; nt < 4; ++nt) {
#pragma unroll
        for (int r = 0; r < 8; ++r) {
            const int m = wave * 16 + r + hi * 8;
            if (m < n_nodes)
                xp[(size_t)m * HC + nt * 16 + m16] = acc[nt][r];
        }
    }
}

// ------------------------------------------------------------------
// Kernel 2: per-edge raw logits + segment max (atomic on sortable uint)
// ------------------------------------------------------------------
__global__ __launch_bounds__(256) void gat_edge_logits(const float* __restrict__ xp,
                                                       const int* __restrict__ ei,
                                                       const float* __restrict__ att,
                                                       float* __restrict__ alpha,
                                                       unsigned* __restrict__ seg_max,
                                                       int n_edges) {
    __shared__ float satt[HC];
    if (threadIdx.x < HC) satt[threadIdx.x] = att[threadIdx.x];
    __syncthreads();

    const int e = blockIdx.x * blockDim.x + threadIdx.x;
    if (e >= n_edges) return;
    const int row = ei[e];
    const int col = ei[n_edges + e];

    const float4* xi4 = (const float4*)(xp + (size_t)row * HC);
    const float4* xj4 = (const float4*)(xp + (size_t)col * HC);

#pragma unroll
    for (int h = 0; h < HEADS; ++h) {
        float s = 0.0f;
#pragma unroll
        for (int q = 0; q < 4; ++q) {
            const float4 a4 = xi4[h * 4 + q];
            const float4 b4 = xj4[h * 4 + q];
            float v0 = a4.x + b4.x, v1 = a4.y + b4.y;
            float v2 = a4.z + b4.z, v3 = a4.w + b4.w;
            v0 = v0 > 0.0f ? v0 : NEG_SLOPE * v0;
            v1 = v1 > 0.0f ? v1 : NEG_SLOPE * v1;
            v2 = v2 > 0.0f ? v2 : NEG_SLOPE * v2;
            v3 = v3 > 0.0f ? v3 : NEG_SLOPE * v3;
            const int c = h * 16 + q * 4;
            s += v0 * satt[c] + v1 * satt[c + 1] + v2 * satt[c + 2] + v3 * satt[c + 3];
        }
        alpha[(size_t)e * HEADS + h] = s;
        atomicMax(&seg_max[(size_t)col * HEADS + h], f2s(s));
    }
}

// ------------------------------------------------------------------
// Kernel 3: ex = exp(alpha - seg_max[col]); seg_sum += ex (atomic f32)
// One thread per (edge, head).
// ------------------------------------------------------------------
__global__ __launch_bounds__(256) void gat_edge_expsum(const int* __restrict__ ei,
                                                       float* __restrict__ alpha,
                                                       const unsigned* __restrict__ seg_max,
                                                       float* __restrict__ seg_sum,
                                                       int n_edges) {
    const int t = blockIdx.x * blockDim.x + threadIdx.x;
    if (t >= n_edges * HEADS) return;
    const int e = t >> 2;
    const int h = t & 3;
    const int col = ei[n_edges + e];
    const float m = s2f(seg_max[(size_t)col * HEADS + h]);
    const float ex = __expf(alpha[t] - m);
    alpha[t] = ex;
    atomicAdd(&seg_sum[(size_t)col * HEADS + h], ex);
}

// ------------------------------------------------------------------
// Kernel 4: weighted scatter-add of x_j directly into the head-mean
// output: out[col][c] += 0.25 * (ex/(sum+eps)) * x_j[h][c]
// One thread per (edge, head).
// ------------------------------------------------------------------
__global__ __launch_bounds__(256) void gat_edge_scatter(const float* __restrict__ xp,
                                                        const int* __restrict__ ei,
                                                        const float* __restrict__ alpha,
                                                        const float* __restrict__ seg_sum,
                                                        float* __restrict__ out,
                                                        int n_edges) {
    const int t = blockIdx.x * blockDim.x + threadIdx.x;
    if (t >= n_edges * HEADS) return;
    const int e = t >> 2;
    const int h = t & 3;
    const int col = ei[n_edges + e];
    const float w = alpha[t] / (seg_sum[(size_t)col * HEADS + h] + EPS) * 0.25f;

    const float4* xj4 = (const float4*)(xp + (size_t)col * HC + h * 16);
    float* o = out + (size_t)col * OUT_CH;
#pragma unroll
    for (int q = 0; q < 4; ++q) {
        const float4 v = xj4[q];
        atomicAdd(&o[q * 4 + 0], v.x * w);
        atomicAdd(&o[q * 4 + 1], v.y * w);
        atomicAdd(&o[q * 4 + 2], v.z * w);
        atomicAdd(&o[q * 4 + 3], v.w * w);
    }
}

// ------------------------------------------------------------------
extern "C" void kernel_launch(void* const* d_in, const int* in_sizes, int n_in,
                              void* d_out, int out_size, void* d_ws, size_t ws_size,
                              hipStream_t stream) {
    const float* x   = (const float*)d_in[0];   // [N, 128]
    const int*   ei  = (const int*)d_in[1];     // [2, E]
    const float* W   = (const float*)d_in[2];   // [128, 64]
    const float* att = (const float*)d_in[3];   // [1, 4, 16]

    const int n_nodes = in_sizes[0] / IN_CH;
    const int n_edges = in_sizes[1] / 2;

    // workspace layout (floats): xp[N*64] | alpha[E*4] | seg_max[N*4] | seg_sum[N*4]
    float*    xp      = (float*)d_ws;
    float*    alpha   = xp + (size_t)n_nodes * HC;
    unsigned* seg_max = (unsigned*)(alpha + (size_t)n_edges * HEADS);
    float*    seg_sum = (float*)(seg_max + (size_t)n_nodes * HEADS);
    float*    out     = (float*)d_out;

    // 0) init
    {
        int threads = n_nodes * OUT_CH;
        gat_init<<<(threads + 255) / 256, 256, 0, stream>>>(out, seg_max, seg_sum, n_nodes);
    }
    // 1) projection GEMM (WMMA bf16, W staged in LDS)
    {
        int n_tiles = (n_nodes + 15) / 16;
        int waves_per_blk = 8;                       // 256 threads = 8 wave32
        int blocks = (n_tiles + waves_per_blk - 1) / waves_per_blk;
        gat_gemm_wmma<<<blocks, 256, 0, stream>>>(x, W, xp, n_nodes, n_tiles);
    }
    // 2) per-edge logits + segment max
    gat_edge_logits<<<(n_edges + 255) / 256, 256, 0, stream>>>(xp, ei, att, alpha,
                                                               seg_max, n_edges);
    // 3) exp + segment sum
    gat_edge_expsum<<<(n_edges * HEADS + 255) / 256, 256, 0, stream>>>(ei, alpha, seg_max,
                                                                      seg_sum, n_edges);
    // 4) normalize + weighted scatter-add (head mean folded in)
    gat_edge_scatter<<<(n_edges * HEADS + 255) / 256, 256, 0, stream>>>(xp, ei, alpha,
                                                                       seg_sum, out, n_edges);
}
